// GraphAttnTrfAggregationModule_28295244546278
// MI455X (gfx1250) — matrix operand order repaired and checked by reading, not compiled, not run on hardware
//
#include <hip/hip_runtime.h>
#include <hip/hip_bf16.h>

// Problem constants (from reference)
#define N_NODES 50000
#define DIM     256
#define NHEAD   8
#define NT      2
#define NEDGE   800000
#define HD      32
#define QKVW    1536   // DIM * NT * 3
#define TSTRIDE 768    // per-t stride inside a qkv row: NHEAD*3*HD

typedef __bf16 bf16x16 __attribute__((ext_vector_type(16)));
typedef float  f32x8   __attribute__((ext_vector_type(8)));

union WFrag {
  bf16x16        v;
  uint4          q[2];
  unsigned int   u[8];
  unsigned short h[16];
};

__device__ __forceinline__ unsigned short f2bf(float f) {
  unsigned u = __float_as_uint(f);
  u += 0x7FFFu + ((u >> 16) & 1u);        // round-to-nearest-even
  return (unsigned short)(u >> 16);
}
__device__ __forceinline__ unsigned f2ord(float f) {   // monotone float->uint
  unsigned u = __float_as_uint(f);
  return (u & 0x80000000u) ? ~u : (u | 0x80000000u);
}
__device__ __forceinline__ float ord2f(unsigned u) {
  u = (u & 0x80000000u) ? (u ^ 0x80000000u) : ~u;
  return __uint_as_float(u);
}

// ---------------- small utility kernels ----------------
__global__ void k_cast_bf16(const float* __restrict__ src,
                            unsigned short* __restrict__ dst, long n) {
  long i = (long)blockIdx.x * blockDim.x + threadIdx.x;
  if (i < n) dst[i] = f2bf(src[i]);
}
__global__ void k_fill_u32(unsigned* __restrict__ p, unsigned v, long n) {
  long i = (long)blockIdx.x * blockDim.x + threadIdx.x;
  if (i < n) p[i] = v;
}
__global__ void k_fill_f32(float* __restrict__ p, float v, long n) {
  long i = (long)blockIdx.x * blockDim.x + threadIdx.x;
  if (i < n) p[i] = v;
}
__global__ void k_copy_x(const float* __restrict__ x, float* __restrict__ out) {
  long i = (long)blockIdx.x * blockDim.x + threadIdx.x;
  if (i < (long)N_NODES * DIM) out[(i >> 8) * (3 * DIM) + (i & 255)] = x[i];
}

// Pack weights (f32, K=256 x ncols row-major) into WMMA B-fragment-major bf16:
// Bp[(((ntile*8 + kt)*32 + lane)*16) + m] = bf16( B[(kt*32 + (lane>>4)*16 + m)*ncols
//                                                  + ntile*16 + (lane&15)] )
// so in the GEMM each lane reads its whole 16-element fragment as two b128 loads,
// consecutive lanes -> consecutive 32B blocks (fully coalesced, streamed).
__global__ void k_pack_b(const float* __restrict__ B,
                         unsigned short* __restrict__ Bp, int ncols) {
  const int i = blockIdx.x * blockDim.x + threadIdx.x;   // total = 256*ncols
  if (i >= 256 * ncols) return;
  const int m     = i & 15;
  const int lane  = (i >> 4) & 31;
  const int kt    = (i >> 9) & 7;
  const int ntile = i >> 12;
  const int col = ntile * 16 + (lane & 15);
  const int k   = kt * 32 + ((lane >> 4) << 4) + m;
  Bp[i] = f2bf(B[(size_t)k * ncols + col]);
}

// ---------------- WMMA bf16 GEMM ----------------
// C[mbase:+16, colBase + ntile*16 :+16] = A(Mx256,bf16) * Bpacked + bias
// Block: 256 threads = 8 waves; grid.x = M/16, grid.y = (#coltiles)/8.
// A strip (16x256 bf16 = 8KB) staged in LDS; B read as packed fragments.
// Inner loop per k-step: 2x ds_load_b128 (A) + 2x global_load_b128 (B) + 1 WMMA.
__global__ void k_gemm_bf16(const unsigned short* __restrict__ A,
                            const unsigned short* __restrict__ Bp,
                            const float* __restrict__ bias,
                            float* __restrict__ C,
                            int ldc, int colBase) {
  __shared__ __align__(16) unsigned short As[16 * 256];
  const int tid   = threadIdx.x;
  const int mbase = blockIdx.x * 16;

  // cooperative 8KB copy of the contiguous 16x256 bf16 A strip (b128 both sides)
  const uint4* Ag = (const uint4*)(A + (size_t)mbase * 256);
  uint4* Al = (uint4*)As;
  Al[tid]       = Ag[tid];
  Al[256 + tid] = Ag[256 + tid];
  __syncthreads();

  const int lane  = tid & 31;
  const int wave  = tid >> 5;
  const int hi    = lane >> 4;     // lane half selects K sub-block (contiguous: k = 16*hi + s, same as packed B)
  const int lrow  = lane & 15;
  const int ntile = blockIdx.y * 8 + wave;
  const int col   = ntile * 16 + lrow;      // output column within this matrix

  // per-lane packed-B stream: fragment f (=kt) lives at +f*32*16 elements
  const uint4* bstream = (const uint4*)(Bp + (((size_t)ntile * 8) * 32 + lane) * 16);

  f32x8 acc = {0.f, 0.f, 0.f, 0.f, 0.f, 0.f, 0.f, 0.f};

#pragma unroll
  for (int kt = 0; kt < 8; ++kt) {          // K = 256 = 8 * 32
    WFrag a, b;
    // A 16x32 bf16 fragment: lanes<16 hold K {0..15}, lanes>=16 {16..31}
    // -> two contiguous 16B chunks per lane, served by ds_load_b128.
    const unsigned short* arow = &As[lrow * 256 + kt * 32 + (hi ? 16 : 0)];
    a.q[0] = *(const uint4*)(arow);
    a.q[1] = *(const uint4*)(arow + 8);
    // packed B fragment: 32 contiguous bytes per lane
    b.q[0] = bstream[kt * 64 + 0];   // 64 uint4 per fragment-block (32 lanes * 2)
    b.q[1] = bstream[kt * 64 + 1];
    acc = __builtin_amdgcn_wmma_f32_16x16x32_bf16(false, a.v, false, b.v,
                                                  (short)0, acc, false, false);
  }

  // C/D 16x16 f32 layout: VGPR r -> row r (lanes 0-15) / row 8+r (lanes 16-31)
  const float bb = bias[col];
#pragma unroll
  for (int r = 0; r < 8; ++r) {
    const int row = mbase + hi * 8 + r;
    C[(size_t)row * ldc + colBase + col] = acc[r] + bb;
  }
}

// ---------------- edge phase ----------------
// one thread per (edge, head): s = scale * <k[src], q[dst]>, segment max via atomicMax
__global__ void k_scores(const float* __restrict__ qkv, const int* __restrict__ esrc,
                         const int* __restrict__ edst, float* __restrict__ sbuf,
                         unsigned* __restrict__ mord) {
  const int t = blockIdx.y;
  const long idx = (long)blockIdx.x * blockDim.x + threadIdx.x;
  if (idx >= (long)NEDGE * NHEAD) return;
  const int e = (int)(idx >> 3);
  const int h = (int)(idx & 7);
  const int src = esrc[(size_t)t * NEDGE + e];
  const int dst = edst[(size_t)t * NEDGE + e];
  const float* kp = qkv + (size_t)src * QKVW + t * TSTRIDE + h * 96 + HD;  // k slot
  const float* qp = qkv + (size_t)dst * QKVW + t * TSTRIDE + h * 96;       // q slot
  float s = 0.f;
#pragma unroll
  for (int d = 0; d < HD; ++d) s += kp[d] * qp[d];
  s *= 0.17677669529663687f;   // 1/sqrt(32)
  sbuf[(size_t)t * NEDGE * NHEAD + idx] = s;
  atomicMax(&mord[((size_t)t * N_NODES + dst) * NHEAD + h], f2ord(s));
}

__global__ void k_expsum(const int* __restrict__ edst, float* __restrict__ sbuf,
                         const unsigned* __restrict__ mord, float* __restrict__ z) {
  const int t = blockIdx.y;
  const long idx = (long)blockIdx.x * blockDim.x + threadIdx.x;
  if (idx >= (long)NEDGE * NHEAD) return;
  const int e = (int)(idx >> 3);
  const int h = (int)(idx & 7);
  const int dst = edst[(size_t)t * NEDGE + e];
  const float m = ord2f(mord[((size_t)t * N_NODES + dst) * NHEAD + h]);
  const float s = sbuf[(size_t)t * NEDGE * NHEAD + idx];
  const float p = __expf(s - m);
  sbuf[(size_t)t * NEDGE * NHEAD + idx] = p;
  atomicAdd(&z[((size_t)t * N_NODES + dst) * NHEAD + h], p);
}

// one block (256 threads) per edge: thread c handles (h = c/32, d = c%32)
__global__ void k_agg(const float* __restrict__ qkv, const int* __restrict__ esrc,
                      const int* __restrict__ edst, const float* __restrict__ sbuf,
                      const float* __restrict__ z, float* __restrict__ agg) {
  const int t = blockIdx.y;
  const int e = blockIdx.x;
  const int c = threadIdx.x;
  const int h = c >> 5;
  const int d = c & 31;
  const int src = esrc[(size_t)t * NEDGE + e];
  const int dst = edst[(size_t)t * NEDGE + e];
  const float p  = sbuf[((size_t)t * NEDGE + e) * NHEAD + h];
  const float zz = z[((size_t)t * N_NODES + dst) * NHEAD + h];
  const float vv = qkv[(size_t)src * QKVW + t * TSTRIDE + h * 96 + 2 * HD + d];  // v slot
  atomicAdd(&agg[((size_t)t * N_NODES + dst) * DIM + c], (p / zz) * vv);
}

// ---------------- launcher ----------------
extern "C" void kernel_launch(void* const* d_in, const int* in_sizes, int n_in,
                              void* d_out, int out_size, void* d_ws, size_t ws_size,
                              hipStream_t stream) {
  (void)in_sizes; (void)n_in; (void)out_size; (void)ws_size;
  const float* x    = (const float*)d_in[0];
  const float* Wqkv = (const float*)d_in[1];
  const float* bqkv = (const float*)d_in[2];
  const float* Wout = (const float*)d_in[3];
  const float* bout = (const float*)d_in[4];
  const int*   esrc = (const int*)d_in[5];
  const int*   edst = (const int*)d_in[6];
  float* out = (float*)d_out;

  char* ws = (char*)d_ws;
  size_t off = 0;
  auto take = [&](size_t bytes) -> void* {
    void* p = ws + off;
    off = (off + bytes + 255) & ~(size_t)255;
    return p;
  };
  unsigned short* xh   = (unsigned short*)take((size_t)N_NODES * DIM * 2);
  unsigned short* wqp  = (unsigned short*)take((size_t)DIM * QKVW * 2);     // packed
  unsigned short* wop  = (unsigned short*)take((size_t)NT * DIM * DIM * 2); // packed
  float*          qkv  = (float*)take((size_t)N_NODES * QKVW * 4);
  unsigned*       mord = (unsigned*)take((size_t)NT * N_NODES * NHEAD * 4);
  float*          zb   = (float*)take((size_t)NT * N_NODES * NHEAD * 4);
  float*          sbuf = (float*)take((size_t)NT * NEDGE * NHEAD * 4);
  float*          agg  = (float*)take((size_t)NT * N_NODES * DIM * 4);
  unsigned short* aggh = (unsigned short*)take((size_t)NT * N_NODES * DIM * 2);

  const int TB = 256;
  auto nb = [](long n) { return (unsigned)((n + 255) / 256); };

  // casts / packs / inits (ws & out are poisoned -> must initialize)
  k_cast_bf16<<<nb((long)N_NODES * DIM), TB, 0, stream>>>(x, xh, (long)N_NODES * DIM);
  k_pack_b<<<nb((long)DIM * QKVW), TB, 0, stream>>>(Wqkv, wqp, QKVW);
  for (int t = 0; t < NT; ++t)
    k_pack_b<<<nb((long)DIM * DIM), TB, 0, stream>>>(
        Wout + (size_t)t * DIM * DIM, wop + (size_t)t * DIM * DIM, DIM);
  k_copy_x<<<nb((long)N_NODES * DIM), TB, 0, stream>>>(x, out);
  k_fill_u32<<<nb((long)NT * N_NODES * NHEAD), TB, 0, stream>>>(
      mord, 0x00800000u /* ord(-FLT_MAX) */, (long)NT * N_NODES * NHEAD);
  k_fill_f32<<<nb((long)NT * N_NODES * NHEAD), TB, 0, stream>>>(
      zb, 0.f, (long)NT * N_NODES * NHEAD);
  k_fill_f32<<<nb((long)NT * N_NODES * DIM), TB, 0, stream>>>(
      agg, 0.f, (long)NT * N_NODES * DIM);

  // qkv = x @ W_qkv + b_qkv   (M=50000=3125*16, K=256, N=1536=12*8*16)
  {
    dim3 g(N_NODES / 16, QKVW / (8 * 16));
    k_gemm_bf16<<<g, 256, 0, stream>>>(xh, wqp, bqkv, qkv, QKVW, 0);
  }

  // per-edge attention (segment softmax + weighted V scatter)
  {
    dim3 g(nb((long)NEDGE * NHEAD), NT);
    k_scores<<<g, 256, 0, stream>>>(qkv, esrc, edst, sbuf, mord);
    k_expsum<<<g, 256, 0, stream>>>(edst, sbuf, mord, zb);
  }
  {
    dim3 g(NEDGE, NT);
    k_agg<<<g, 256, 0, stream>>>(qkv, esrc, edst, sbuf, zb, agg);
  }

  // out_t = agg_t @ W_out[t] + b_out[t]  -> out[:, 256 + t*256 : 256 + (t+1)*256]
  k_cast_bf16<<<nb((long)NT * N_NODES * DIM), TB, 0, stream>>>(
      agg, aggh, (long)NT * N_NODES * DIM);
  for (int t = 0; t < NT; ++t) {
    dim3 g(N_NODES / 16, DIM / (8 * 16));
    k_gemm_bf16<<<g, 256, 0, stream>>>(aggh + (size_t)t * N_NODES * DIM,
                                       wop + (size_t)t * DIM * DIM,
                                       bout + (size_t)t * DIM,
                                       out, 3 * DIM, DIM + t * DIM);
  }
}
